// CCPM_65867618452131
// MI455X (gfx1250) — compile-verified, matrix operand order checked by reading, not run on hardware
//
#include <hip/hip_runtime.h>
#include <hip/hip_bf16.h>

// ---------------------------------------------------------------------------
// Model constants (from reference)
// ---------------------------------------------------------------------------
#define BATCH 4096
#define NFLD  26
#define VOC   100000
#define EDIM  64
#define NDNS  13
#define DIN   3085          // 3*16*64 + 13
#define KPAD  3104          // DIN padded to multiple of 32
#define N1    256
#define N2    128
#define WT    4             // w-tile width per CNN block (64/WT = 16 tiles)

typedef __attribute__((ext_vector_type(16))) __bf16 v16bf;
typedef __attribute__((ext_vector_type(8)))  float  v8f;

union U16x16 { unsigned int u[8]; v16bf v; };

__device__ inline unsigned short f32_to_bf16(float f) {
    unsigned int u = __float_as_uint(f);
    unsigned int r = (u + 0x7FFFu + ((u >> 16) & 1u)) >> 16;   // RNE
    return (unsigned short)r;
}
__device__ inline float bf16_to_f32(unsigned short h) {
    return __uint_as_float(((unsigned int)h) << 16);
}

// Inline tanh via hardware v_exp_f32 + fast divide (no OCML call).
__device__ inline float fast_tanh(float x) {
    return 1.0f - __fdividef(2.0f, __expf(2.0f * x) + 1.0f);
}

// Sorted (descending) top-k; tie-break = lowest index (matches jax.lax.top_k).
template <int NIN, int NOUT>
__device__ inline void topk_desc(const float* y, float* out) {
    unsigned int used = 0u;
    #pragma unroll
    for (int t = 0; t < NOUT; ++t) {
        float best = -3.4e38f; int bi = 0;
        #pragma unroll
        for (int h = 0; h < NIN; ++h) {
            bool ok = ((used >> h) & 1u) == 0u && y[h] > best;
            best = ok ? y[h] : best;
            bi   = ok ? h    : bi;
        }
        used |= (1u << bi);
        out[t] = best;
    }
}

// ---------------------------------------------------------------------------
// Weight conversion kernels
// ---------------------------------------------------------------------------
// f32 (K x N row-major, K zero-padded to Kp) -> bf16 pair-major dst[(k/2)*N+n][2]
__global__ void cvt_w_pairs(const float* __restrict__ src, unsigned short* __restrict__ dst,
                            int Kp, int N, int Kreal) {
    int i = blockIdx.x * blockDim.x + threadIdx.x;
    if (i >= Kp * N) return;
    int k = i / N, n = i % N;
    float v = (k < Kreal) ? src[(size_t)k * N + n] : 0.0f;
    dst[(((size_t)(k >> 1) * N + n) << 1) | (size_t)(k & 1)] = f32_to_bf16(v);
}

__global__ void cvt_plain(const float* __restrict__ src, unsigned short* __restrict__ dst, int n) {
    int i = blockIdx.x * blockDim.x + threadIdx.x;
    if (i < n) dst[i] = f32_to_bf16(src[i]);
}

__global__ void fill_dense(const float* __restrict__ dense, unsigned short* __restrict__ flatA) {
    int b = blockIdx.x, j = threadIdx.x;     // 32 threads
    float v = (j < NDNS) ? dense[b * NDNS + j] : 0.0f;
    flatA[(size_t)b * KPAD + 3072 + j] = f32_to_bf16(v);
}

// ---------------------------------------------------------------------------
// Fused CNN. One block per (b, w-tile of 4).
// Phase 1: gather -> conv1+tanh+top23 scattered DIRECTLY into a pair-major
//          im2col matrix s_Bp[160][96][2] (zero-filled halo = conv padding).
// Phase 2: conv2 = WMMA GEMM (A from global bf16 weights, B from s_Bp; every
//          fragment reg is ONE load with an immediate offset), accumulators
//          kept in registers across a barrier, then epilogue overlays s_Bp
//          with s_o2/s_x2p/conv3 weights -> top8 -> conv3+top3 -> flat.
// ---------------------------------------------------------------------------
__global__ __launch_bounds__(256) void cnn_fused(
    const int*   __restrict__ sparse, const float* __restrict__ emb,
    const float* __restrict__ cw1, const float* __restrict__ cb1,
    const unsigned short* __restrict__ A2g,  // conv2 weights bf16 [32][320]
    const float* __restrict__ cb2,
    const float* __restrict__ cw3, const float* __restrict__ cb3,
    unsigned short* __restrict__ flatA)
{
    __shared__ __align__(16) unsigned char s_big[61440];
    __shared__ float s_x0p[32][WT];          // row = f + 3, halo = 0
    __shared__ float s_cw1[64 * 7];
    __shared__ float s_cb1[64];
    __shared__ float s_cb2[32];

    unsigned short* s_Bp  = (unsigned short*)s_big;             // ph1: [160][96][2]
    unsigned short* s_o2  = (unsigned short*)s_big;             // ph2: [32][24][4]  (6144 B)
    float*          s_x2p = (float*)(s_big + 6144);             // ph2: [32][10][4]  (5120 B)
    float*          s_cw3 = (float*)(s_big + 11264);            // ph2: 1536 f32     (6144 B)
    float*          s_cb3 = (float*)(s_big + 17408);            // ph2: 16 f32

    const int b   = blockIdx.x >> 4;
    const int wt  = blockIdx.x & 15;
    const int tid = threadIdx.x;

    // --- phase-1 staging ---------------------------------------------------
    {
        unsigned int* z = (unsigned int*)s_big;                 // zero im2col (halo = conv pad)
        for (int i = tid; i < 15360; i += 256) z[i] = 0u;
        float* z0 = &s_x0p[0][0];
        for (int i = tid; i < 32 * WT; i += 256) z0[i] = 0.0f;
    }
    for (int i = tid; i < 64 * 7; i += 256) s_cw1[i] = cw1[i];
    if (tid < 64) s_cb1[tid] = cb1[tid];
    if (tid < 32) s_cb2[tid] = cb2[tid];

    for (int i = tid; i < NFLD * WT; i += 256) {                // embedding gather
        int f = i / WT, w = i % WT;
        int idx = sparse[b * NFLD + f];
        s_x0p[f + 3][w] = emb[((size_t)f * VOC + (size_t)idx) * EDIM + wt * WT + w];
    }
    __syncthreads();

    // --- conv1 + tanh + top23, scattered into im2col -----------------------
    {
        const int ch = tid >> 2, w = tid & 3;    // 64*4 = 256 tasks
        float y[26];
        #pragma unroll
        for (int h = 0; h < 26; ++h) {
            float a = s_cb1[ch];
            #pragma unroll
            for (int t = 0; t < 7; ++t)
                a += s_cw1[ch * 7 + t] * s_x0p[h + t][w];
            y[h] = fast_tanh(a);
        }
        float o[23];
        topk_desc<26, 23>(y, o);
        // scatter: x1[ch][t][w] -> B[k=5ch+dk][n=(t+2-dk)*4+w] for valid h
        #pragma unroll
        for (int t = 0; t < 23; ++t) {
            unsigned short bv = f32_to_bf16(o[t]);
            #pragma unroll
            for (int dk = 0; dk < 5; ++dk) {
                int h = t + 2 - dk;
                if (h >= 0 && h < 24) {
                    int k = ch * 5 + dk;
                    s_Bp[((k >> 1) * 96 + (h * 4 + w)) * 2 + (k & 1)] = bv;
                }
            }
        }
    }
    __syncthreads();

    // --- conv2 WMMA GEMM: M=32, K=320, N=96 (h*4+w; h=23 col = scratch) ----
    const int wave = tid >> 5, lane = tid & 31;
    const int half = lane >> 4, l15 = lane & 15;
    v8f acc0, acc1;
    const bool has2 = (wave < 4);               // waves 0-3 own a second tile

    auto run_tile = [&](int tile) -> v8f {
        const int mt = tile / 6, nt = tile % 6;
        const int n  = nt * 16 + l15;
        const unsigned int* Ab = (const unsigned int*)(A2g + (mt * 16 + l15) * 320);
        const unsigned int* Bb = (const unsigned int*)(s_Bp + n * 2);
        v8f acc = {0.f, 0.f, 0.f, 0.f, 0.f, 0.f, 0.f, 0.f};
        #pragma unroll
        for (int kk = 0; kk < 320; kk += 32) {
            U16x16 Af, Bf;
            #pragma unroll
            for (int v = 0; v < 8; ++v) {
                const int kb = ((v < 4) ? 0 : 16) + half * 8 + (v & 3) * 2;
                const int p  = (kk + kb) >> 1;          // pair index; imm offsets
                Af.u[v] = Ab[p];
                Bf.u[v] = Bb[p * 96];
            }
            acc = __builtin_amdgcn_wmma_f32_16x16x32_bf16(
                      false, Af.v, false, Bf.v, (short)0, acc, false, false);
        }
        return acc;
    };
    acc0 = run_tile(wave);
    if (has2) acc1 = run_tile(wave + 8);        // wave-uniform branch
    __syncthreads();                            // all s_Bp reads retired

    // --- phase-2 staging (overlays s_Bp) + conv2 epilogue ------------------
    for (int i = tid; i < 16 * 32 * 3; i += 256) s_cw3[i] = cw3[i];
    if (tid < 16) s_cb3[tid] = cb3[tid];
    for (int i = tid; i < 32 * 10 * WT; i += 256) s_x2p[i] = 0.0f;  // halo = 0

    auto store_tile = [&](const v8f& acc, int tile) {
        const int mt = tile / 6, nt = tile % 6;
        const int n  = nt * 16 + l15;
        const int h  = n >> 2, w = n & 3;       // h in [0,23]; row 23 = scratch
        #pragma unroll
        for (int r = 0; r < 8; ++r) {
            const int oc = mt * 16 + r + 8 * half;
            s_o2[(oc * 24 + h) * 4 + w] = f32_to_bf16(fast_tanh(acc[r] + s_cb2[oc]));
        }
    };
    store_tile(acc0, wave);
    if (has2) store_tile(acc1, wave + 8);
    __syncthreads();

    // --- top8 over H=23 ----------------------------------------------------
    if (tid < 128) {
        const int oc = tid >> 2, w = tid & 3;
        float y[23];
        #pragma unroll
        for (int h = 0; h < 23; ++h) y[h] = bf16_to_f32(s_o2[(oc * 24 + h) * 4 + w]);
        float o[8];
        topk_desc<23, 8>(y, o);
        #pragma unroll
        for (int t = 0; t < 8; ++t) s_x2p[(oc * 10 + t + 1) * 4 + w] = o[t];
    }
    __syncthreads();

    // --- conv3 (32->16, k=3 over H=8) + tanh + top3 -> flat ---------------
    if (tid < 64) {
        const int oc = tid >> 2, w = tid & 3;
        float y[8];
        for (int h = 0; h < 8; ++h) {
            float a = s_cb3[oc];
            for (int ic = 0; ic < 32; ++ic) {
                #pragma unroll
                for (int dk = 0; dk < 3; ++dk)
                    a += s_cw3[(oc * 32 + ic) * 3 + dk] * s_x2p[(ic * 10 + h + dk) * 4 + w];
            }
            y[h] = fast_tanh(a);
        }
        float o[3];
        topk_desc<8, 3>(y, o);
        const int wg = wt * WT + w;
        #pragma unroll
        for (int t = 0; t < 3; ++t)
            flatA[(size_t)b * KPAD + (oc * 3 + t) * 64 + wg] = f32_to_bf16(o[t]);
    }
}

// ---------------------------------------------------------------------------
// Generic bf16 WMMA GEMM: C = A @ B + bias.  A row-major (k-pairs contiguous),
// B pair-major [(k/2)*N+n][2]; one wave per 16x16 tile; K%32==0.
// ---------------------------------------------------------------------------
__global__ __launch_bounds__(256) void gemm_bf16_wmma(
    const unsigned short* __restrict__ A, const unsigned short* __restrict__ B,
    const float* __restrict__ bias, float* __restrict__ C,
    int M, int N, int K)
{
    const int lane = threadIdx.x & 31;
    const int wave = threadIdx.x >> 5;
    const int tile = blockIdx.x * 8 + wave;
    const int ntn  = N >> 4;
    const int mt   = tile / ntn, nt = tile % ntn;
    if (mt >= (M >> 4)) return;                 // wave-uniform exit
    const int half = lane >> 4, l15 = lane & 15;
    const int row  = mt * 16 + l15;
    const int col  = nt * 16 + l15;
    v8f acc = {0.f, 0.f, 0.f, 0.f, 0.f, 0.f, 0.f, 0.f};
    #pragma unroll 2
    for (int kk = 0; kk < K; kk += 32) {
        U16x16 Af, Bf;
        #pragma unroll
        for (int v = 0; v < 8; ++v) {
            const int kb = ((v < 4) ? 0 : 16) + half * 8 + (v & 3) * 2;
            const int k0 = kk + kb;
            Af.u[v] = *(const unsigned int*)(A + (size_t)row * K + k0);
            Bf.u[v] = *(const unsigned int*)(B + (((size_t)(k0 >> 1) * N + col) << 1));
        }
        acc = __builtin_amdgcn_wmma_f32_16x16x32_bf16(
                  false, Af.v, false, Bf.v, (short)0, acc, false, false);
    }
    #pragma unroll
    for (int r = 0; r < 8; ++r) {
        const int m = mt * 16 + r + 8 * half;
        C[(size_t)m * N + col] = acc[r] + bias[col];
    }
}

// ---------------------------------------------------------------------------
// BatchNorm (population stats over M rows) + tanh. One block per column.
// ---------------------------------------------------------------------------
__global__ __launch_bounds__(256) void bn_tanh(
    const float* __restrict__ X, const float* __restrict__ g, const float* __restrict__ be,
    void* __restrict__ out, int M, int N, int out_bf16)
{
    const int col = blockIdx.x;
    __shared__ float s1[256], s2[256];
    float s = 0.f, sq = 0.f;
    for (int r = threadIdx.x; r < M; r += 256) {
        float v = X[(size_t)r * N + col];
        s += v; sq += v * v;
    }
    s1[threadIdx.x] = s; s2[threadIdx.x] = sq;
    __syncthreads();
    for (int off = 128; off > 0; off >>= 1) {
        if (threadIdx.x < off) {
            s1[threadIdx.x] += s1[threadIdx.x + off];
            s2[threadIdx.x] += s2[threadIdx.x + off];
        }
        __syncthreads();
    }
    const float mu  = s1[0] / (float)M;
    const float var = s2[0] / (float)M - mu * mu;
    const float inv = rsqrtf(var + 1e-5f);
    const float gg = g[col], bb = be[col];
    for (int r = threadIdx.x; r < M; r += 256) {
        float v = fast_tanh(gg * (X[(size_t)r * N + col] - mu) * inv + bb);
        if (out_bf16) ((unsigned short*)out)[(size_t)r * N + col] = f32_to_bf16(v);
        else          ((float*)out)[(size_t)r * N + col] = v;
    }
}

// ---------------------------------------------------------------------------
__global__ void final_dot(const float* __restrict__ H, const float* __restrict__ W3,
                          const float* __restrict__ b3, float* __restrict__ out)
{
    int row = blockIdx.x * blockDim.x + threadIdx.x;
    if (row < BATCH) {
        float a = b3[0];
        #pragma unroll 8
        for (int j = 0; j < N2; ++j) a += H[(size_t)row * N2 + j] * W3[j];
        out[row] = a;
    }
}

// ---------------------------------------------------------------------------
extern "C" void kernel_launch(void* const* d_in, const int* in_sizes, int n_in,
                              void* d_out, int out_size, void* d_ws, size_t ws_size,
                              hipStream_t stream)
{
    (void)in_sizes; (void)n_in; (void)out_size; (void)ws_size;
    const int*   sparse = (const int*)  d_in[0];
    const float* dense  = (const float*)d_in[1];
    const float* emb    = (const float*)d_in[2];
    const float* cw1    = (const float*)d_in[3];
    const float* cb1    = (const float*)d_in[4];
    const float* cw2    = (const float*)d_in[5];
    const float* cb2    = (const float*)d_in[6];
    const float* cw3    = (const float*)d_in[7];
    const float* cb3    = (const float*)d_in[8];
    const float* W1     = (const float*)d_in[9];
    const float* b1     = (const float*)d_in[10];
    const float* g1     = (const float*)d_in[11];
    const float* be1    = (const float*)d_in[12];
    const float* W2     = (const float*)d_in[13];
    const float* b2     = (const float*)d_in[14];
    const float* g2     = (const float*)d_in[15];
    const float* be2    = (const float*)d_in[16];
    const float* W3     = (const float*)d_in[17];
    const float* b3     = (const float*)d_in[18];

    char* ws = (char*)d_ws;
    size_t off = 0;
    auto take = [&](size_t bytes) -> char* {
        char* p = ws + off;
        off = (off + bytes + 255) & ~(size_t)255;
        return p;
    };
    unsigned short* flatA = (unsigned short*)take((size_t)BATCH * KPAD * 2);
    unsigned short* W1b   = (unsigned short*)take((size_t)KPAD * N1 * 2);   // pair-major
    float*          H1    = (float*)         take((size_t)BATCH * N1 * 4);
    unsigned short* h1b   = (unsigned short*)take((size_t)BATCH * N1 * 2);
    unsigned short* W2b   = (unsigned short*)take((size_t)N1 * N2 * 2);     // pair-major
    float*          H2    = (float*)         take((size_t)BATCH * N2 * 4);
    float*          h2f   = (float*)         take((size_t)BATCH * N2 * 4);
    unsigned short* A2b   = (unsigned short*)take((size_t)32 * 320 * 2);    // conv2 W bf16

    // weight prep
    cvt_w_pairs<<<(KPAD * N1 + 255) / 256, 256, 0, stream>>>(W1, W1b, KPAD, N1, DIN);
    cvt_w_pairs<<<(N1 * N2 + 255) / 256, 256, 0, stream>>>(W2, W2b, N1, N2, N1);
    cvt_plain  <<<(32 * 320 + 255) / 256, 256, 0, stream>>>(cw2, A2b, 32 * 320);
    fill_dense<<<BATCH, 32, 0, stream>>>(dense, flatA);

    // fused CNN stack (WMMA conv2)
    cnn_fused<<<BATCH * 16, 256, 0, stream>>>(sparse, emb, cw1, cb1, A2b, cb2, cw3, cb3, flatA);

    // MLP layer 1: (4096 x 3104) @ (3104 x 256)
    gemm_bf16_wmma<<<512, 256, 0, stream>>>(flatA, W1b, b1, H1, BATCH, N1, KPAD);
    bn_tanh<<<N1, 256, 0, stream>>>(H1, g1, be1, (void*)h1b, BATCH, N1, 1);

    // MLP layer 2: (4096 x 256) @ (256 x 128)
    gemm_bf16_wmma<<<256, 256, 0, stream>>>(h1b, W2b, b2, H2, BATCH, N2, N1);
    bn_tanh<<<N2, 256, 0, stream>>>(H2, g2, be2, (void*)h2f, BATCH, N2, 0);

    // output head
    final_dot<<<16, 256, 0, stream>>>(h2f, W3, b3, (float*)d_out);
}
